// ChannelAdaptivePolarQuant_76381698392698
// MI455X (gfx1250) — matrix operand order repaired and checked by reading, not compiled, not run on hardware
//
#include <hip/hip_runtime.h>

#define B_ROWS 262144
#define DDIM   128
#define NH     32
#define NL     96
#define KH     16
#define KL     4
#define NTILES (B_ROWS / 16)

typedef __attribute__((ext_vector_type(16))) __bf16 v16bf;
typedef __attribute__((ext_vector_type(8)))  __bf16 v8bf;
typedef __attribute__((ext_vector_type(4)))  __bf16 v4bf;
typedef __attribute__((ext_vector_type(8)))  float  v8f;

union AFrag { v16bf v; v8bf h[2]; };

__device__ __forceinline__ v8f wmma_bf16(v16bf a, v16bf b, v8f c) {
  return __builtin_amdgcn_wmma_f32_16x16x32_bf16(
      false, a, false, b, (short)0, c, false, false);
}

__device__ __forceinline__ unsigned pack_hl(float c) {
  __bf16 h = (__bf16)c;
  __bf16 l = (__bf16)(c - (float)h);
  return (unsigned)__builtin_bit_cast(unsigned short, h) |
         ((unsigned)__builtin_bit_cast(unsigned short, l) << 16);
}

__global__ __launch_bounds__(256) void capq_kernel(
    const float* __restrict__ x, const float* __restrict__ Pi,
    const float* __restrict__ cHiG, const float* __restrict__ cLoG,
    const int* __restrict__ hiIdx, const int* __restrict__ loIdx,
    float* __restrict__ xhat, int* __restrict__ outHi, int* __restrict__ outLo)
{
  // bf16 hi/lo tiles; row stride 136 (272B = 17*16B: keeps 16B alignment, staggers banks)
  __shared__ __attribute__((aligned(16))) __bf16 xH[16][136];
  __shared__ __attribute__((aligned(16))) __bf16 xL[16][136];
  __shared__ __attribute__((aligned(16))) __bf16 yH[16][136];
  __shared__ __attribute__((aligned(16))) __bf16 yL[16][136];
  __shared__ unsigned pkHi[KH], pkLo[KL];   // centroid -> packed (hi,lo) bf16

  const int tid  = threadIdx.x;
  const int w    = tid >> 5;        // wave 0..7: owns permuted N-tile (GEMM1) / d-tile (GEMM2)
  const int lane = tid & 31;
  const int row  = lane & 15;
  const int half = lane >> 4;

  if (tid < KH) pkHi[tid] = pack_hl(cHiG[tid]);
  if (tid < KL) pkLo[tid] = pack_hl(cLoG[tid]);

  // Permuted column space: columns 0..31 = high channels (sorted), 32..127 = low channels.
  const int np   = w * 16 + row;                         // permuted column this lane owns
  const int chan = (w < 2) ? hiIdx[np] : loIdx[np - NH]; // actual channel index in k-space

  // Midpoint thresholds in registers (wave-uniform choice of path).
  float thrH[KH - 1], thrL[KL - 1];
  if (w < 2) {
    #pragma unroll
    for (int i = 0; i < KH - 1; ++i) thrH[i] = 0.5f * (cHiG[i] + cHiG[i + 1]);
  } else {
    #pragma unroll
    for (int i = 0; i < KL - 1; ++i) thrL[i] = 0.5f * (cLoG[i] + cLoG[i + 1]);
  }

  // One-time register-resident Pi B-fragments (hi/lo split).
  // GEMM1 (permuted N): B[K][N=row] = Pi[chan][K]
  // GEMM2 (natural d):  B[K][N=row] = Pi[K][w*16+row]
  v16bf b1h[4], b1l[4], b2h[4], b2l[4];
  #pragma unroll
  for (int kc = 0; kc < 4; ++kc) {
    #pragma unroll
    for (int j = 0; j < 16; ++j) {
      const int kk = kc * 32 + 8 * half + (j & 7) + 16 * (j >> 3); // B-layout: elem j -> K = 8*half + (j&7) + 16*(j>>3)
      float p1 = Pi[chan * DDIM + kk];
      __bf16 h1 = (__bf16)p1;
      b1h[kc][j] = h1;
      b1l[kc][j] = (__bf16)(p1 - (float)h1);
      float p2 = Pi[kk * DDIM + w * 16 + row];
      __bf16 h2 = (__bf16)p2;
      b2h[kc][j] = h2;
      b2l[kc][j] = (__bf16)(p2 - (float)h2);
    }
  }
  __syncthreads();

  // Per-lane 32-bit store offsets, computed once; per-r strides fold into the
  // instruction's immediate offset (uniform 64-bit base advances per tile).
  const unsigned xoff = (unsigned)(8 * half * DDIM + w * 16 + row);
  const unsigned hoff = (unsigned)(8 * half * NH + np);
  const unsigned loff = (unsigned)(8 * half * NL + (np - NH));

  for (int tile = blockIdx.x; tile < NTILES; tile += gridDim.x) {
    const float* src  = x     + (size_t)tile * (16 * DDIM);
    float*       xout = xhat  + (size_t)tile * (16 * DDIM);
    int*         hout = outHi + (size_t)tile * (16 * NH);
    int*         lout = outLo + (size_t)tile * (16 * NL);

    // ---- coalesced load + single fp32->bf16 hi/lo conversion into LDS ----
    #pragma unroll
    for (int c = 0; c < 2; ++c) {
      unsigned fi = (unsigned)(tid + c * 256) * 4;       // 2048 floats total
      float4 v = *(const float4*)(src + fi);
      int r = fi >> 7, cc = fi & 127;
      __bf16 h0 = (__bf16)v.x, h1 = (__bf16)v.y, h2 = (__bf16)v.z, h3 = (__bf16)v.w;
      v4bf hv = {h0, h1, h2, h3};
      v4bf lv = {(__bf16)(v.x - (float)h0), (__bf16)(v.y - (float)h1),
                 (__bf16)(v.z - (float)h2), (__bf16)(v.w - (float)h3)};
      *(v4bf*)&xH[r][cc] = hv;
      *(v4bf*)&xL[r][cc] = lv;
    }
    __syncthreads();

    // ---- GEMM1: y[:,perm] = x * Pi[perm,:]^T  (A frags = pure LDS b128 loads) ----
    v8f acc = {};
    #pragma unroll
    for (int kc = 0; kc < 4; ++kc) {
      const int off = kc * 32 + 8 * half;                // A runs [off..off+7], [off+16..off+23]
      AFrag ah, al;
      ah.h[0] = *(const v8bf*)&xH[row][off];
      ah.h[1] = *(const v8bf*)&xH[row][off + 16];
      al.h[0] = *(const v8bf*)&xL[row][off];
      al.h[1] = *(const v8bf*)&xL[row][off + 16];
      acc = wmma_bf16(ah.v, b1h[kc], acc);
      acc = wmma_bf16(al.v, b1h[kc], acc);
      acc = wmma_bf16(ah.v, b1l[kc], acc);
    }

    // ---- quantize (wave-uniform high/low path), emit idx, write packed bf16 y_hat ----
    if (w < 2) {
      #pragma unroll
      for (int r = 0; r < 8; ++r) {
        float y = acc[r];
        int idx = 0;
        #pragma unroll
        for (int i = 0; i < KH - 1; ++i) idx += (y > thrH[i]) ? 1 : 0;
        hout[hoff + r * NH] = idx;                       // C-layout row M = r + 8*half
        unsigned p = pkHi[idx];
        yH[r + 8 * half][chan] = __builtin_bit_cast(__bf16, (unsigned short)(p & 0xffffu));
        yL[r + 8 * half][chan] = __builtin_bit_cast(__bf16, (unsigned short)(p >> 16));
      }
    } else {
      #pragma unroll
      for (int r = 0; r < 8; ++r) {
        float y = acc[r];
        int idx = 0;
        #pragma unroll
        for (int i = 0; i < KL - 1; ++i) idx += (y > thrL[i]) ? 1 : 0;
        lout[loff + r * NL] = idx;
        unsigned p = pkLo[idx];
        yH[r + 8 * half][chan] = __builtin_bit_cast(__bf16, (unsigned short)(p & 0xffffu));
        yL[r + 8 * half][chan] = __builtin_bit_cast(__bf16, (unsigned short)(p >> 16));
      }
    }
    __syncthreads();

    // ---- GEMM2: x_hat = y_hat * Pi  (A frags = pure LDS b128 loads, no conversions) ----
    v8f acc2 = {};
    #pragma unroll
    for (int kc = 0; kc < 4; ++kc) {
      const int off = kc * 32 + 8 * half;
      AFrag ah, al;
      ah.h[0] = *(const v8bf*)&yH[row][off];
      ah.h[1] = *(const v8bf*)&yH[row][off + 16];
      al.h[0] = *(const v8bf*)&yL[row][off];
      al.h[1] = *(const v8bf*)&yL[row][off + 16];
      acc2 = wmma_bf16(ah.v, b2h[kc], acc2);
      acc2 = wmma_bf16(al.v, b2h[kc], acc2);
      acc2 = wmma_bf16(ah.v, b2l[kc], acc2);
    }
    #pragma unroll
    for (int r = 0; r < 8; ++r)
      xout[xoff + r * DDIM] = acc2[r];
    __syncthreads();
  }
}

extern "C" void kernel_launch(void* const* d_in, const int* in_sizes, int n_in,
                              void* d_out, int out_size, void* d_ws, size_t ws_size,
                              hipStream_t stream) {
  const float* x   = (const float*)d_in[0];
  const float* Pi  = (const float*)d_in[1];
  const float* cHi = (const float*)d_in[2];
  const float* cLo = (const float*)d_in[3];
  const int*   hiI = (const int*)d_in[4];
  const int*   loI = (const int*)d_in[5];

  float* xhat = (float*)d_out;
  int*   oHi  = (int*)((float*)d_out + (size_t)B_ROWS * DDIM);
  int*   oLo  = oHi + (size_t)B_ROWS * NH;

  capq_kernel<<<1024, 256, 0, stream>>>(x, Pi, cHi, cLo, hiI, loI, xhat, oHi, oLo);
}
